// AttentionGRPE_57320633533120
// MI455X (gfx1250) — matrix operand order repaired and checked by reading, not compiled, not run on hardware
//
#include <hip/hip_runtime.h>

// ---------------------------------------------------------------------------
// AttentionGRPE for MI455X (gfx1250, wave32, WMMA).
// Mixed precision: fp32 inputs -> f16 operands, fp32 WMMA accumulate.
// All matrix math through v_wmma_f32_16x16x32_f16; 16x64 register blocking.
// ---------------------------------------------------------------------------

typedef __attribute__((ext_vector_type(16))) _Float16 v16h;
typedef __attribute__((ext_vector_type(8)))  _Float16 v8h;
typedef __attribute__((ext_vector_type(4)))  _Float16 v4h;
typedef __attribute__((ext_vector_type(8)))  float    v8f;

#define BATCH  8
#define NTOK   1024
#define HEADS  8
#define DHEAD  64
#define DIM    512
#define INNER  512          // HEADS * DHEAD
#define QKVW   1536         // 3 * INNER
#define SCALE  0.125f       // DHEAD^-0.5

// ---- WMMA fragment loaders (layouts per CDNA5 ISA 7.12.2) ------------------
// A (16x32 f16): lane<16 holds row M=lane, K {k0..k0+7, k0+16..k0+23};
//                lane>=16 holds row M=lane-16, K {k0+8..k0+15, k0+24..k0+31}.
__device__ __forceinline__ v16h load_frag_a(const _Float16* __restrict__ base,
                                            int stride, int k0, int lane) {
  int r  = lane & 15;
  int kk = (lane < 16) ? 0 : 8;
  const _Float16* p = base + r * stride + k0 + kk;
  union { v16h v; v8h h[2]; } u;
  u.h[0] = *(const v8h*)(p);        // 16B aligned
  u.h[1] = *(const v8h*)(p + 16);
  return u.v;
}

// B (32x16 f16) from transposed storage BT[n][k]:
// lane<16 holds col N=lane, K {k0..k0+7, k0+16..k0+23}; lane>=16 col N=lane-16, K {k0+8..k0+15, k0+24..k0+31}.
__device__ __forceinline__ v16h load_frag_b(const _Float16* __restrict__ baseT,
                                            int stride, int k0, int lane) {
  int c  = lane & 15;
  int kk = (lane < 16) ? 0 : 8;
  const _Float16* p = baseT + c * stride + k0 + kk;
  union { v16h v; v8h h[2]; } u;
  u.h[0] = *(const v8h*)(p);
  u.h[1] = *(const v8h*)(p + 16);
  return u.v;
}

__device__ __forceinline__ v8f wmma_f16(v16h a, v16h b, v8f c) {
  return __builtin_amdgcn_wmma_f32_16x16x32_f16(false, a, false, b,
                                                (short)0, c, false, false);
}

// ---- precision conversion kernels ------------------------------------------
__global__ void k_f32_to_f16(const float* __restrict__ src,
                             _Float16* __restrict__ dst, int n) {
  int i = (blockIdx.x * blockDim.x + threadIdx.x) * 4;
  if (i < n) {
    float4 f = *(const float4*)(src + i);
    v4h h = { (_Float16)f.x, (_Float16)f.y, (_Float16)f.z, (_Float16)f.w };
    *(v4h*)(dst + i) = h;
  }
}

// src [rows][cols] fp32 -> dst [cols][rows] f16 (transposed for B-fragments)
__global__ void k_f32_to_f16_T(const float* __restrict__ src,
                               _Float16* __restrict__ dst, int rows, int cols) {
  int i = blockIdx.x * blockDim.x + threadIdx.x;
  if (i < rows * cols) {
    int r = i / cols, c = i % cols;
    dst[c * rows + r] = (_Float16)src[i];
  }
}

// ---- QKV projection: [8192,512] @ WqkvT -> per-head q / k / vT (f16) -------
// 1 wave per 16x64 tile (4 accumulators); 512*24 tiles; 64 WMMAs per wave.
// A fragment reused across 4 column tiles per K-step.
__global__ void __launch_bounds__(256)
k_qkv_gemm(const _Float16* __restrict__ X, const _Float16* __restrict__ WT,
           _Float16* __restrict__ qb, _Float16* __restrict__ kb,
           _Float16* __restrict__ vT) {
  int lane = threadIdx.x & 31;
  int wave = threadIdx.x >> 5;
  int tile = blockIdx.x * 8 + wave;          // 12288 tiles total
  int tm = tile / 24, tn = tile % 24;
  int m0 = tm * 16, n0 = tn * 64;
  const _Float16* a_base = X  + (size_t)m0 * DIM;
  const _Float16* b_base = WT + (size_t)n0 * DIM;
  v8f acc[4] = {};
  for (int k0 = 0; k0 < DIM; k0 += 32) {
    __builtin_prefetch(a_base + ((lane & 15) * DIM) + k0 + 64, 0, 1);
    v16h a = load_frag_a(a_base, DIM, k0, lane);
#pragma unroll
    for (int t = 0; t < 4; ++t) {
      v16h b = load_frag_b(b_base + (size_t)(t * 16) * DIM, DIM, k0, lane);
      acc[t] = wmma_f16(a, b, acc[t]);
    }
  }
  // 64-wide tile lives entirely in one (segment, head): seg = tn/8, h = tn%8
  int c  = lane & 15;
  int r0 = (lane < 16) ? 0 : 8;
  int seg = tn >> 3;                         // 0=q 1=k 2=v
  int h   = tn & 7;
#pragma unroll
  for (int t = 0; t < 4; ++t) {
    int d = t * 16 + c;
#pragma unroll
    for (int r = 0; r < 8; ++r) {
      int row = m0 + r0 + r;
      int bi = row >> 10, tok = row & 1023;
      int bh = bi * HEADS + h;
      _Float16 val = (_Float16)acc[t][r];
      if (seg == 0)      qb[((size_t)bh * NTOK + tok) * DHEAD + d] = val;
      else if (seg == 1) kb[((size_t)bh * NTOK + tok) * DHEAD + d] = val;
      else               vT[((size_t)bh * DHEAD + d) * NTOK + tok] = val;
    }
  }
}

// ---- fused attention block: 16 rows x 1024 cols per workgroup --------------
// Phase 1: raw scores via WMMA (q-fragments hoisted, 4-tile register block)
// Phase 2: softmax(dots0) streamed to out2 (the BW-dominant 268MB output)
// Phase 3: in-place dots = s*scale + rel_bias[REL_IDX] + 0.01*exp(-factor*DIS)
// Phase 4: softmax(dots) -> f16 attn in LDS
// Phase 5: attn @ V via WMMA; all 8 waves (K split), LDS partial reduction
__global__ void __launch_bounds__(256)
k_attention(const _Float16* __restrict__ qb, const _Float16* __restrict__ kb,
            const _Float16* __restrict__ vT, const float* __restrict__ rel_table,
            const float* __restrict__ headsita, const int* __restrict__ rpe,
            _Float16* __restrict__ attnOut, float* __restrict__ out2) {
  extern __shared__ char smem[];
  float*    sS = (float*)smem;                        // [16][1024] fp32 scores
  _Float16* sA = (_Float16*)(smem + 16 * NTOK * 4);   // [16][1024] f16 attn

  int tid = threadIdx.x, lane = tid & 31, wave = tid >> 5;
  int i0 = blockIdx.x * 16;                  // row-tile of tokens
  int bh = blockIdx.y;                       // batch*head
  int b  = bh >> 3, h = bh & 7;
  const _Float16* qbase = qb + ((size_t)bh * NTOK + i0) * DHEAD;
  const _Float16* kbase = kb + (size_t)bh * NTOK * DHEAD;
  const _Float16* vbase = vT + (size_t)bh * DHEAD * NTOK;

  // --- Phase 1: s = q k^T (raw, unscaled); q A-fragments loaded once ---
  v16h qa0 = load_frag_a(qbase, DHEAD, 0,  lane);
  v16h qa1 = load_frag_a(qbase, DHEAD, 32, lane);
  int c = lane & 15, r0 = (lane < 16) ? 0 : 8;
#pragma unroll
  for (int g = 0; g < 2; ++g) {              // each wave owns a 128-col strip
    int j0 = wave * 128 + g * 64;
    v8f acc[4] = {};
#pragma unroll
    for (int t = 0; t < 4; ++t) {
      const _Float16* kt = kbase + (size_t)(j0 + t * 16) * DHEAD;
      acc[t] = wmma_f16(qa0, load_frag_b(kt, DHEAD, 0,  lane), acc[t]);
      acc[t] = wmma_f16(qa1, load_frag_b(kt, DHEAD, 32, lane), acc[t]);
    }
#pragma unroll
    for (int t = 0; t < 4; ++t)
#pragma unroll
      for (int r = 0; r < 8; ++r)
        sS[(r0 + r) * NTOK + (j0 + t * 16 + c)] = acc[t][r];
  }
  __syncthreads();

  // --- Phase 2: softmax(dots0) -> out2 (one wave owns each row) ---
  for (int i = wave; i < 16; i += 8) {
    const float* rowp = sS + i * NTOK;
    float m = -1e30f;
    for (int j = lane; j < NTOK; j += 32) m = fmaxf(m, rowp[j] * SCALE);
#pragma unroll
    for (int off = 16; off; off >>= 1) m = fmaxf(m, __shfl_xor(m, off, 32));
    float s = 0.f;
    for (int j = lane; j < NTOK; j += 32) s += __expf(rowp[j] * SCALE - m);
#pragma unroll
    for (int off = 16; off; off >>= 1) s += __shfl_xor(s, off, 32);
    float inv = 1.f / s;
    float* orow = out2 + ((size_t)bh * NTOK + (i0 + i)) * NTOK;
    for (int j = lane; j < NTOK; j += 32)
      orow[j] = __expf(rowp[j] * SCALE - m) * inv;     // coalesced 128B/wave
  }
  __syncthreads();

  // --- Phase 3: dots = s*scale + bias + 0.01*pos (REL_IDX/DIS analytic) ---
  int use_rpe = rpe[0];
  float th = headsita[h];
  float factor = 1.f / (2.f * th * th + 1e-10f);
  for (int e = tid; e < 16 * NTOK; e += 256) {
    int i = e >> 10, j = e & 1023;
    float s0 = sS[e] * SCALE;
    if (use_rpe) {
      int ti = i0 + i;
      int dr = (ti >> 5) - (j >> 5);                   // row delta in 32x32 grid
      int dc = (ti & 31) - (j & 31);                   // col delta
      float dis = (float)(dr * dr + dc * dc) * (1.0f / 1024.0f);
      int idx = (dr + 31) * 63 + (dc + 31);            // REL_IDX
      s0 += rel_table[idx * HEADS + h] + 0.01f * __expf(-factor * dis);
    }
    sS[e] = s0;
  }
  __syncthreads();

  // --- Phase 4: softmax(dots) -> f16 attn in LDS ---
  for (int i = wave; i < 16; i += 8) {
    const float* rowp = sS + i * NTOK;
    float m = -1e30f;
    for (int j = lane; j < NTOK; j += 32) m = fmaxf(m, rowp[j]);
#pragma unroll
    for (int off = 16; off; off >>= 1) m = fmaxf(m, __shfl_xor(m, off, 32));
    float s = 0.f;
    for (int j = lane; j < NTOK; j += 32) s += __expf(rowp[j] - m);
#pragma unroll
    for (int off = 16; off; off >>= 1) s += __shfl_xor(s, off, 32);
    float inv = 1.f / s;
    for (int j = lane; j < NTOK; j += 32)
      sA[i * NTOK + j] = (_Float16)(__expf(rowp[j] - m) * inv);
  }
  __syncthreads();

  // --- Phase 5: out = attn @ V (16 x 64); 8 waves: d-tile = wave&3, K-half ---
  {
    int n0 = (wave & 3) * 16;
    int kh = (wave >> 2) * 512;              // K half: [0,512) or [512,1024)
    v8f acc = {};
    for (int k0 = kh; k0 < kh + 512; k0 += 32) {
      v16h a  = load_frag_a(sA, NTOK, k0, lane);       // ds_load_b128 from LDS
      v16h bf = load_frag_b(vbase + (size_t)n0 * NTOK, NTOK, k0, lane);
      acc = wmma_f16(a, bf, acc);
    }
    float* sP = sS;                          // score region is free now
    if (wave >= 4) {
#pragma unroll
      for (int r = 0; r < 8; ++r) sP[(wave - 4) * 256 + lane * 8 + r] = acc[r];
    }
    __syncthreads();
    if (wave < 4) {
#pragma unroll
      for (int r = 0; r < 8; ++r) {
        float v = acc[r] + sP[wave * 256 + lane * 8 + r];
        int t = i0 + r0 + r;
        attnOut[((size_t)b * NTOK + t) * INNER + h * DHEAD + n0 + c] =
            (_Float16)v;
      }
    }
  }
}

// ---- output projection: attnOut[8192,512] @ WoutT + b_out -> fp32 ----------
// 1 wave per 16x64 tile; 512*8 tiles.
__global__ void __launch_bounds__(256)
k_out_gemm(const _Float16* __restrict__ A, const _Float16* __restrict__ WT,
           const float* __restrict__ bias, float* __restrict__ out) {
  int lane = threadIdx.x & 31;
  int wave = threadIdx.x >> 5;
  int tile = blockIdx.x * 8 + wave;          // 4096 tiles total
  int tm = tile >> 3, tn = tile & 7;
  int m0 = tm * 16, n0 = tn * 64;
  const _Float16* a_base = A  + (size_t)m0 * DIM;
  const _Float16* b_base = WT + (size_t)n0 * DIM;
  v8f acc[4] = {};
  for (int k0 = 0; k0 < DIM; k0 += 32) {
    v16h a = load_frag_a(a_base, DIM, k0, lane);
#pragma unroll
    for (int t = 0; t < 4; ++t) {
      v16h b = load_frag_b(b_base + (size_t)(t * 16) * DIM, DIM, k0, lane);
      acc[t] = wmma_f16(a, b, acc[t]);
    }
  }
  int c = lane & 15, r0 = (lane < 16) ? 0 : 8;
#pragma unroll
  for (int t = 0; t < 4; ++t) {
    int col = n0 + t * 16 + c;
    float bv = bias[col];
#pragma unroll
    for (int r = 0; r < 8; ++r)
      out[(size_t)(m0 + r0 + r) * DIM + col] = acc[t][r] + bv;
  }
}

// ---------------------------------------------------------------------------
extern "C" void kernel_launch(void* const* d_in, const int* in_sizes, int n_in,
                              void* d_out, int out_size, void* d_ws, size_t ws_size,
                              hipStream_t stream) {
  const float* x         = (const float*)d_in[0];  // [8,1024,512]
  const float* W_qkv     = (const float*)d_in[1];  // [512,1536]
  const float* W_out     = (const float*)d_in[2];  // [512,512]
  const float* b_out     = (const float*)d_in[3];  // [512]
  const float* rel_table = (const float*)d_in[4];  // [3969,8]
  const float* headsita  = (const float*)d_in[5];  // [8]
  const int*   rpe       = (const int*)d_in[6];    // [1]

  // workspace carve-up (f16), ~44 MB total
  _Float16* Xbf   = (_Float16*)d_ws;
  _Float16* WqkvT = Xbf   + (size_t)8192 * 512;
  _Float16* WoutT = WqkvT + (size_t)QKVW * 512;
  _Float16* qb    = WoutT + (size_t)512 * 512;
  _Float16* kb    = qb    + (size_t)BATCH * HEADS * NTOK * DHEAD;
  _Float16* vT    = kb    + (size_t)BATCH * HEADS * NTOK * DHEAD;
  _Float16* aOut  = vT    + (size_t)BATCH * HEADS * NTOK * DHEAD;

  float* out1 = (float*)d_out;                         // [8,1024,512]
  float* out2 = out1 + (size_t)BATCH * NTOK * INNER;   // [8,8,1024,1024]

  k_f32_to_f16<<<(8192 * 512) / (256 * 4), 256, 0, stream>>>(x, Xbf, 8192 * 512);
  k_f32_to_f16_T<<<(512 * QKVW + 255) / 256, 256, 0, stream>>>(W_qkv, WqkvT, 512, QKVW);
  k_f32_to_f16_T<<<(512 * 512 + 255) / 256, 256, 0, stream>>>(W_out, WoutT, 512, 512);

  k_qkv_gemm<<<1536, 256, 0, stream>>>(Xbf, WqkvT, qb, kb, vT);

  dim3 ag(NTOK / 16, BATCH * HEADS);                   // (64, 64)
  size_t lds = (size_t)16 * NTOK * 4 + (size_t)16 * NTOK * 2;  // 96 KB dynamic
  k_attention<<<ag, 256, lds, stream>>>(qb, kb, vT, rel_table, headsita, rpe,
                                        aOut, out2);

  k_out_gemm<<<512, 256, 0, stream>>>(aOut, WoutT, b_out, out1);
}
